// AERGCN_77678778515475
// MI455X (gfx1250) — matrix operand order repaired and checked by reading, not compiled, not run on hardware
//
#include <hip/hip_runtime.h>
#include <hip/hip_bf16.h>

// ---------------- problem constants (from reference) ----------------
static constexpr int B_   = 8;
static constexpr int S_   = 128;
static constexpr int F_   = 768;   // F_IN == F_OUT
static constexpr int R_   = 41;    // NUM_REL
static constexpr int NH_  = 8;
static constexpr int HID_ = 96;
static constexpr int EMB_ = 768;

typedef __bf16 bf16_t;
typedef __attribute__((ext_vector_type(16))) __bf16 v16bf;
typedef __attribute__((ext_vector_type(8)))  __bf16 v8bf;
typedef __attribute__((ext_vector_type(8)))  float  v8f;

// ---------------- WMMA fragment helpers (wave32, 16x16x32 bf16) ----------------
// A (16x32) from row-major MxK memory. Per-lane data is two contiguous
// 16-byte runs: K in [8*half, 8*half+8) and [16+8*half, 16+8*half+8).
__device__ __forceinline__ v16bf load_a16(const bf16_t* __restrict__ p, int ld) {
  const int lane = threadIdx.x & 31;
  const int m = lane & 15, half = lane >> 4;
  const bf16_t* q = p + (long)m * ld + 8 * half;
  v8bf lo = *(const v8bf*)q;          // k = 8h .. 8h+7
  v8bf hi = *(const v8bf*)(q + 16);   // k = 16+8h .. 16+8h+7
  return __builtin_shufflevector(lo, hi, 0, 1, 2, 3, 4, 5, 6, 7,
                                 8, 9, 10, 11, 12, 13, 14, 15);
}

// B (32x16) from TRANSPOSED NxK memory (ld = K stride). Per-lane data is two
// contiguous 16-byte runs: K in [8*half, 8*half+8) and [16+8*half, 16+8*half+8).
__device__ __forceinline__ v16bf load_b16_nk(const bf16_t* __restrict__ p, int ld) {
  const int lane = threadIdx.x & 31;
  const int n = lane & 15, half = lane >> 4;
  const bf16_t* q = p + (long)n * ld + 8 * half;
  v8bf lo = *(const v8bf*)q;
  v8bf hi = *(const v8bf*)(q + 16);
  return __builtin_shufflevector(lo, hi, 0, 1, 2, 3, 4, 5, 6, 7,
                                 8, 9, 10, 11, 12, 13, 14, 15);
}

__device__ __forceinline__ v8f wmma_bf16(v16bf a, v16bf b, v8f c) {
  return __builtin_amdgcn_wmma_f32_16x16x32_bf16(false, a, false, b, (short)0, c,
                                                 false, false);
}

// C/D 16x16 f32: VGPR i -> row i + 8*half, col = lane&15
__device__ __forceinline__ void store_c_bf16(bf16_t* dst, int ld, v8f c) {
  const int lane = threadIdx.x & 31;
  const int n = lane & 15, half = lane >> 4;
#pragma unroll
  for (int i = 0; i < 8; ++i) dst[(long)(i + 8 * half) * ld + n] = (bf16_t)c[i];
}

__device__ __forceinline__ float wave_sum(float v) {
#pragma unroll
  for (int off = 16; off; off >>= 1) v += __shfl_xor(v, off, 32);
  return v;
}

// ---------------- conversion kernels ----------------
// out[m][c][r] = (bf16) in[m][r][c]   (transpose + downconvert, nmat batches)
__global__ void cvt_transpose_bf16(const float* __restrict__ in,
                                   bf16_t* __restrict__ out,
                                   int rows, int cols, long nmat) {
  long idx = (long)blockIdx.x * blockDim.x + threadIdx.x;
  const long stride = (long)gridDim.x * blockDim.x;
  const long per = (long)rows * cols;
  const long total = nmat * per;
  for (; idx < total; idx += stride) {
    const long m = idx / per;
    const long rc = idx % per;
    const int r = (int)(rc / cols), c = (int)(rc % cols);
    out[m * per + (long)c * rows + r] = (bf16_t)in[idx];
  }
}

// ---------------- reduction / elementwise kernels ----------------
// rcp[(b*R+r)*S + s] = 1/sum_k adj  (1 if sum==0)
__global__ void denom_rcp_kernel(const float* __restrict__ adj,
                                 float* __restrict__ rcp, int rows) {
  const int wid = threadIdx.x >> 5, lane = threadIdx.x & 31;
  const int row = blockIdx.x * (blockDim.x >> 5) + wid;
  if (row >= rows) return;
  const float* p = adj + (long)row * S_;
  float s = 0.f;
#pragma unroll
  for (int j = 0; j < S_ / 32; ++j) s += p[lane + 32 * j];
  s = wave_sum(s);
  if (lane == 0) rcp[row] = (s == 0.f) ? 1.f : 1.f / s;
}

// u[r*F+i] = sum_o w[r,i,o] * sw[o]
__global__ void u_kernel(const float* __restrict__ w, const float* __restrict__ sw,
                         float* __restrict__ u, int rows) {
  const int wid = threadIdx.x >> 5, lane = threadIdx.x & 31;
  const int row = blockIdx.x * (blockDim.x >> 5) + wid;
  if (row >= rows) return;
  const float* p = w + (long)row * F_;
  float s = 0.f;
  for (int o = lane; o < F_; o += 32) s += p[o] * sw[o];
  s = wave_sum(s);
  if (lane == 0) u[row] = s;
}

// t[(b*R+r)*S + k] = dot(h[b,k,:], u[r,:])
__global__ void t_kernel(const float* __restrict__ h, const float* __restrict__ u,
                         float* __restrict__ t, int rows) {
  const int wid = threadIdx.x >> 5, lane = threadIdx.x & 31;
  const int row = blockIdx.x * (blockDim.x >> 5) + wid;
  if (row >= rows) return;
  const int b = row / (R_ * S_);
  const int rk = row % (R_ * S_);
  const int r = rk / S_, k = rk % S_;
  const float* hp = h + ((long)b * S_ + k) * F_;
  const float* up = u + (long)r * F_;
  float s = 0.f;
  for (int i = lane; i < F_; i += 32) s += hp[i] * up[i];
  s = wave_sum(s);
  if (lane == 0) t[row] = s;
}

// logits[(b*R+r)*S + s] = rcp * sum_k adj[b,r,s,k] * t[(b*R+r)*S + k] + sb
__global__ void logits_kernel(const float* __restrict__ adj,
                              const float* __restrict__ t,
                              const float* __restrict__ rcp,
                              const float* __restrict__ sb,
                              float* __restrict__ logits, int rows) {
  const int wid = threadIdx.x >> 5, lane = threadIdx.x & 31;
  const int row = blockIdx.x * (blockDim.x >> 5) + wid;
  if (row >= rows) return;
  const int br = row / S_;
  const float* ap = adj + (long)row * S_;
  const float* tp = t + (long)br * S_;
  float s = 0.f;
#pragma unroll
  for (int j = 0; j < S_ / 32; ++j) s += ap[lane + 32 * j] * tp[lane + 32 * j];
  s = wave_sum(s);
  if (lane == 0) logits[row] = s * rcp[row] + sb[0];
}

// softmax over r per (b,s); cscale = softmax * rcp
__global__ void rel_softmax_kernel(const float* __restrict__ logits,
                                   const float* __restrict__ rcp,
                                   float* __restrict__ cscale) {
  const int idx = blockIdx.x * blockDim.x + threadIdx.x;
  if (idx >= B_ * S_) return;
  const int b = idx / S_, s = idx % S_;
  float m = -1e30f;
  for (int r = 0; r < R_; ++r)
    m = fmaxf(m, logits[(long)(b * R_ + r) * S_ + s]);
  float sum = 0.f;
  for (int r = 0; r < R_; ++r)
    sum += __expf(logits[(long)(b * R_ + r) * S_ + s] - m);
  const float inv = 1.f / sum;
  for (int r = 0; r < R_; ++r) {
    const long o = (long)(b * R_ + r) * S_ + s;
    cscale[o] = __expf(logits[o] - m) * inv * rcp[o];
  }
}

// sadj = bf16( adj * cscale[row] )
__global__ void sadj_kernel(const float* __restrict__ adj,
                            const float* __restrict__ cscale,
                            bf16_t* __restrict__ sadj, long n) {
  long i = (long)blockIdx.x * blockDim.x + threadIdx.x;
  const long stride = (long)gridDim.x * blockDim.x;
  for (; i < n; i += stride) sadj[i] = (bf16_t)(adj[i] * cscale[i / S_]);
}

// ---------------- WMMA kernels ----------------
// g[r][(b,s)][i] = (diag(cscale)·adj)[b,r] @ h[b]   (328 GEMMs of 128x768x128)
// B operand (h) is pre-transposed: ht[b][i][s]  (N x K, ld = S_)
__global__ void gr_gemm_kernel(const bf16_t* __restrict__ sadj,
                               const bf16_t* __restrict__ ht,
                               bf16_t* __restrict__ g) {
  const int z = blockIdx.y;            // b*R + r
  const int b = z / R_, r = z % R_;
  const int wid = threadIdx.x >> 5;
  const int tile = blockIdx.x * 8 + wid;   // 0..47 (4 x 12 tiles of 32x64)
  const int tm = tile / 12, tn = tile % 12;
  const bf16_t* A = sadj + (long)z * S_ * S_ + (long)(tm * 32) * S_;
  const bf16_t* Bt = ht + (long)b * F_ * S_ + (long)(tn * 64) * S_;
  v8f acc[2][4];
#pragma unroll
  for (int im = 0; im < 2; ++im)
#pragma unroll
    for (int j = 0; j < 4; ++j) acc[im][j] = {};
  for (int kk = 0; kk < S_; kk += 32) {
    v16bf a0 = load_a16(A + kk, S_);
    v16bf a1 = load_a16(A + 16 * S_ + kk, S_);
    v16bf bf[4];
#pragma unroll
    for (int j = 0; j < 4; ++j) bf[j] = load_b16_nk(Bt + (long)(16 * j) * S_ + kk, S_);
#pragma unroll
    for (int j = 0; j < 4; ++j) {
      acc[0][j] = wmma_bf16(a0, bf[j], acc[0][j]);
      acc[1][j] = wmma_bf16(a1, bf[j], acc[1][j]);
    }
  }
  bf16_t* out = g + (long)r * (B_ * S_ * F_) + (long)b * S_ * F_;
#pragma unroll
  for (int im = 0; im < 2; ++im)
#pragma unroll
    for (int j = 0; j < 4; ++j)
      store_c_bf16(out + (long)(tm * 32 + 16 * im) * F_ + tn * 64 + 16 * j, F_,
                   acc[im][j]);
}

// out[MxN] = act( sum_{cr} A_cr(MxK) @ Bt_cr(NxK)^T + bias )
// Epilogue: fp32 row-major (outf), optional bf16 row-major (outb),
// optional bf16 transposed-per-batch (outbT[b][col][s], M == B_*S_).
__global__ void wmma_gemm_kernel(const bf16_t* __restrict__ A,
                                 const bf16_t* __restrict__ Bt,
                                 const float* __restrict__ bias,
                                 float* __restrict__ outf,
                                 bf16_t* __restrict__ outb,
                                 bf16_t* __restrict__ outbT,
                                 int M, int N, int K, int num_r,
                                 long a_stride, long b_stride, int relu) {
  const int wid = threadIdx.x >> 5;
  const int tiles_n = N / 64;
  const int tile = blockIdx.x * (blockDim.x >> 5) + wid;
  if (tile >= (M / 32) * tiles_n) return;
  const int tm = tile / tiles_n, tn = tile % tiles_n;
  v8f acc[2][4];
#pragma unroll
  for (int im = 0; im < 2; ++im)
#pragma unroll
    for (int j = 0; j < 4; ++j) acc[im][j] = {};
  for (int cr = 0; cr < num_r; ++cr) {
    const bf16_t* Ar = A + cr * a_stride + (long)(tm * 32) * K;
    const bf16_t* Br = Bt + cr * b_stride + (long)(tn * 64) * K;
    for (int kk = 0; kk < K; kk += 32) {
      if (kk + 32 < K) {  // stream-prefetch next k-step (global_prefetch_b8)
        __builtin_prefetch(Ar + kk + 32, 0, 1);
        __builtin_prefetch(Br + kk + 32, 0, 1);
      }
      v16bf a0 = load_a16(Ar + kk, K);
      v16bf a1 = load_a16(Ar + 16 * K + kk, K);
      v16bf bf[4];
#pragma unroll
      for (int j = 0; j < 4; ++j) bf[j] = load_b16_nk(Br + (long)(16 * j) * K + kk, K);
#pragma unroll
      for (int j = 0; j < 4; ++j) {
        acc[0][j] = wmma_bf16(a0, bf[j], acc[0][j]);
        acc[1][j] = wmma_bf16(a1, bf[j], acc[1][j]);
      }
    }
  }
  const int lane = threadIdx.x & 31;
  const int n = lane & 15, half = lane >> 4;
  auto emit = [&](v8f c, int row0, int col0) {
#pragma unroll
    for (int i = 0; i < 8; ++i) {
      const int row = row0 + i + 8 * half;
      const int col = col0 + n;
      float v = c[i];
      if (bias) v += bias[col];
      if (relu) v = fmaxf(v, 0.f);
      outf[(long)row * N + col] = v;
      if (outb) outb[(long)row * N + col] = (bf16_t)v;
      if (outbT)  // [b][col][s], b = row/S_, s = row%S_
        outbT[(long)(row >> 7) * ((long)N * S_) + (long)col * S_ + (row & 127)] =
            (bf16_t)v;
    }
  };
#pragma unroll
  for (int im = 0; im < 2; ++im)
#pragma unroll
    for (int j = 0; j < 4; ++j)
      emit(acc[im][j], tm * 32 + 16 * im, tn * 64 + 16 * j);
}

// ---------------- attention tail (scalar, tiny) ----------------
__global__ void qx_kernel(const float* __restrict__ q, const float* __restrict__ wq,
                          const float* __restrict__ bq, float* __restrict__ qx) {
  const int idx = blockIdx.x * blockDim.x + threadIdx.x;
  if (idx >= B_ * EMB_) return;
  const int b = idx / EMB_, j = idx % EMB_;
  float s = bq[j];
  for (int i = 0; i < EMB_; ++i) s += q[(long)b * EMB_ + i] * wq[(long)i * EMB_ + j];
  qx[idx] = s;
}

__global__ void qw_kernel(const float* __restrict__ qx, const float* __restrict__ wbil,
                          float* __restrict__ qw) {
  const int idx = blockIdx.x * blockDim.x + threadIdx.x;
  if (idx >= B_ * NH_ * HID_) return;
  const int e = idx % HID_;
  const int bh = idx / HID_;
  float s = 0.f;
  for (int d = 0; d < HID_; ++d) s += qx[(long)bh * HID_ + d] * wbil[(long)d * HID_ + e];
  qw[idx] = s;
}

__global__ void attn_kernel(const float* __restrict__ kx, const float* __restrict__ qw,
                            float* __restrict__ o) {
  __shared__ float qs[HID_];
  __shared__ float sc[S_];
  __shared__ float red[2];
  const int bh = blockIdx.x;
  const int b = bh / NH_, h = bh % NH_;
  const int tid = threadIdx.x;
  if (tid < HID_) qs[tid] = qw[(long)bh * HID_ + tid];
  __syncthreads();
  const float* kbase = kx + (long)b * S_ * EMB_ + h * HID_;
  float s = 0.f;
  for (int e = 0; e < HID_; ++e) s += qs[e] * kbase[(long)tid * EMB_ + e];
  sc[tid] = s;
  __syncthreads();
  if (tid == 0) {
    float m = sc[0];
    for (int i = 1; i < S_; ++i) m = fmaxf(m, sc[i]);
    red[0] = m;
  }
  __syncthreads();
  sc[tid] = __expf(sc[tid] - red[0]);
  __syncthreads();
  if (tid == 0) {
    float sum = 0.f;
    for (int i = 0; i < S_; ++i) sum += sc[i];
    red[1] = 1.f / sum;
  }
  __syncthreads();
  sc[tid] *= red[1];
  __syncthreads();
  if (tid < HID_) {
    float acc = 0.f;
    for (int sI = 0; sI < S_; ++sI) acc += sc[sI] * kbase[(long)sI * EMB_ + tid];
    o[(long)b * EMB_ + h * HID_ + tid] = acc;
  }
}

__global__ void proj_kernel(const float* __restrict__ o, const float* __restrict__ wproj,
                            const float* __restrict__ bproj, float* __restrict__ out) {
  const int idx = blockIdx.x * blockDim.x + threadIdx.x;
  if (idx >= B_ * EMB_) return;
  const int b = idx / EMB_, e = idx % EMB_;
  float s = bproj[e];
  for (int j = 0; j < EMB_; ++j) s += o[(long)b * EMB_ + j] * wproj[(long)j * EMB_ + e];
  out[idx] = s;
}

// ---------------- host launcher ----------------
static inline size_t alignup256(size_t x) { return (x + 255) & ~(size_t)255; }

extern "C" void kernel_launch(void* const* d_in, const int* in_sizes, int n_in,
                              void* d_out, int out_size, void* d_ws, size_t ws_size,
                              hipStream_t stream) {
  const float* x       = (const float*)d_in[0];
  const float* adj     = (const float*)d_in[1];
  const float* q       = (const float*)d_in[2];
  const float* w_rgcn  = (const float*)d_in[3];
  const float* score_w = (const float*)d_in[4];
  const float* score_b = (const float*)d_in[5];
  const float* wk      = (const float*)d_in[6];
  const float* bk      = (const float*)d_in[7];
  const float* wq      = (const float*)d_in[8];
  const float* bq      = (const float*)d_in[9];
  const float* wbil    = (const float*)d_in[10];
  const float* wproj   = (const float*)d_in[11];
  const float* bproj   = (const float*)d_in[12];
  float* out = (float*)d_out;
  (void)in_sizes; (void)n_in; (void)out_size; (void)ws_size;

  char* ws = (char*)d_ws;
  size_t off = 0;
  auto alloc = [&](size_t bytes) -> void* {
    void* p = ws + off;
    off = alignup256(off + bytes);
    return p;
  };

  bf16_t* wtbf  = (bf16_t*)alloc((size_t)2 * R_ * F_ * F_ * 2);  // W^T bf16, ~97 MB
  bf16_t* wktbf = (bf16_t*)alloc((size_t)F_ * F_ * 2);           // wk^T bf16
  bf16_t* ht0   = (bf16_t*)alloc((size_t)B_ * F_ * S_ * 2);      // x^T per batch
  bf16_t* ht1   = (bf16_t*)alloc((size_t)B_ * F_ * S_ * 2);      // h1^T per batch
  bf16_t* hbf2  = (bf16_t*)alloc((size_t)B_ * S_ * F_ * 2);      // h2 row-major
  float*  h1    = (float*)alloc((size_t)B_ * S_ * F_ * 4);
  float*  h2    = (float*)alloc((size_t)B_ * S_ * F_ * 4);
  float*  rcp   = (float*)alloc((size_t)B_ * R_ * S_ * 4);
  float*  u     = (float*)alloc((size_t)R_ * F_ * 4);
  float*  t     = (float*)alloc((size_t)B_ * R_ * S_ * 4);
  float*  lg    = (float*)alloc((size_t)B_ * R_ * S_ * 4);
  float*  cs    = (float*)alloc((size_t)B_ * R_ * S_ * 4);
  bf16_t* sadj  = (bf16_t*)alloc((size_t)B_ * R_ * S_ * S_ * 2); // ~11 MB
  bf16_t* g     = (bf16_t*)alloc((size_t)R_ * B_ * S_ * F_ * 2); // ~64 MB
  float*  kx    = (float*)alloc((size_t)B_ * S_ * F_ * 4);
  float*  qx    = (float*)alloc((size_t)B_ * EMB_ * 4);
  float*  qwv   = (float*)alloc((size_t)B_ * NH_ * HID_ * 4);
  float*  ov    = (float*)alloc((size_t)B_ * EMB_ * 4);

  const int ROWS = B_ * R_ * S_;  // 41984

  // One-time operand preparation: bf16 + transpose so every WMMA B-fragment
  // load is a contiguous 128-bit load.
  cvt_transpose_bf16<<<2048, 256, 0, stream>>>(w_rgcn, wtbf, F_, F_, 2L * R_);
  cvt_transpose_bf16<<<256, 256, 0, stream>>>(wk, wktbf, F_, F_, 1);
  cvt_transpose_bf16<<<256, 256, 0, stream>>>(x, ht0, S_, F_, B_);
  denom_rcp_kernel<<<ROWS / 8, 256, 0, stream>>>(adj, rcp, ROWS);

  for (int l = 0; l < 2; ++l) {
    const float*  hin  = (l == 0) ? x : h1;
    const bf16_t* htin = (l == 0) ? ht0 : ht1;
    float*        hout = (l == 0) ? h1 : h2;
    const float*  wl   = w_rgcn + (size_t)l * R_ * F_ * F_;
    const bf16_t* wlb  = wtbf + (size_t)l * R_ * F_ * F_;

    u_kernel<<<(R_ * F_) / 8, 256, 0, stream>>>(wl, score_w + (size_t)l * F_, u, R_ * F_);
    t_kernel<<<ROWS / 8, 256, 0, stream>>>(hin, u, t, ROWS);
    logits_kernel<<<ROWS / 8, 256, 0, stream>>>(adj, t, rcp, score_b + l, lg, ROWS);
    rel_softmax_kernel<<<(B_ * S_ + 255) / 256, 256, 0, stream>>>(lg, rcp, cs);
    sadj_kernel<<<2048, 256, 0, stream>>>(adj, cs, sadj, (long)ROWS * S_);
    gr_gemm_kernel<<<dim3(6, B_ * R_), 256, 0, stream>>>(sadj, htin, g);
    // layer 0 feeds gr_gemm of layer 1 -> transposed bf16 out;
    // layer 1 feeds the kx GEMM -> row-major bf16 out.
    wmma_gemm_kernel<<<48, 256, 0, stream>>>(
        g, wlb, nullptr, hout, (l == 1) ? hbf2 : nullptr, (l == 0) ? ht1 : nullptr,
        B_ * S_, F_, F_, R_, (long)B_ * S_ * F_, (long)F_ * F_, 1);
  }

  // kx = h @ wk + bk
  wmma_gemm_kernel<<<48, 256, 0, stream>>>(hbf2, wktbf, bk, kx, nullptr, nullptr,
                                           B_ * S_, F_, F_, 1, 0, 0, 0);
  qx_kernel<<<(B_ * EMB_ + 255) / 256, 256, 0, stream>>>(q, wq, bq, qx);
  qw_kernel<<<(B_ * NH_ * HID_ + 255) / 256, 256, 0, stream>>>(qx, wbil, qwv);
  attn_kernel<<<B_ * NH_, S_, 0, stream>>>(kx, qwv, ov);
  proj_kernel<<<(B_ * EMB_ + 255) / 256, 256, 0, stream>>>(ov, wproj, bproj, out);
}
